// DeepSeekV3TokenChoiceTopKRouter_18442589569147
// MI455X (gfx1250) — compile-verified, matrix operand order checked by reading, not run on hardware
//
#include <hip/hip_runtime.h>
#include <hip/hip_bf16.h>

typedef __attribute__((ext_vector_type(16))) __bf16       v16bf;
typedef __attribute__((ext_vector_type(4)))  __bf16       v4bf;
typedef __attribute__((ext_vector_type(8)))  float        v8f;
typedef __attribute__((ext_vector_type(8)))  unsigned int v8u;

#define N_TOK     8192
#define N_EXP     256
#define KDIM      7168
#define NGROUPS   8
#define GROUP_SZ  32
#define TOPK_GRP  4
#define TOP_K     8
#define SCALING   2.5f

#define WAVES        4
#define TOK_PER_BLK  32                        // 2 M-tiles of 16 tokens
#define KSTAGE       64                        // K panel per barrier pair (2 wmma K-steps)
#define STRIDE       72                        // ushorts per panel row (144B, bank-conflict-free)

#define LDS_B_BYTES  (N_EXP * STRIDE * 2)          // 36864: 256 x 64 bf16 (+pad)
#define LDS_A_BYTES  (TOK_PER_BLK * STRIDE * 2)    //  4608:  32 x 64 bf16 (+pad)
#define SMEM_BYTES   (LDS_B_BYTES + LDS_A_BYTES)   // 41472 B arena (reused by epilogue)

// ---------------- Kernel 1: gate_w fp32 -> bf16 into workspace ----------------
__global__ void convert_w_kernel(const float* __restrict__ w,
                                 unsigned short* __restrict__ o) {
    int i = (blockIdx.x * blockDim.x + threadIdx.x) * 4;
    float4 f = *(const float4*)(w + i);
    v4bf pk;
    pk[0] = (__bf16)f.x; pk[1] = (__bf16)f.y;
    pk[2] = (__bf16)f.z; pk[3] = (__bf16)f.w;
    *(uint2*)(o + i) = __builtin_bit_cast(uint2, pk);
}

// Fragment load from an LDS panel (ISA 16-bit A/B striping: per-lane chunks at
// off and off+16; lanes 0-15 use koff=0, lanes 16-31 koff=8 -- folded into off)
static __device__ __forceinline__ v8u loadFrag(const unsigned short* lds,
                                               int tile, int lrow, int off) {
    const unsigned short* p = lds + (tile * 16 + lrow) * STRIDE + off;
    uint4 c0 = *(const uint4*)(p);         // K[off .. off+7]
    uint4 c1 = *(const uint4*)(p + 16);    // K[off+16 .. off+23]
    v8u r;
    r[0] = c0.x; r[1] = c0.y; r[2] = c0.z; r[3] = c0.w;
    r[4] = c1.x; r[5] = c1.y; r[6] = c1.z; r[7] = c1.w;
    return r;
}

// ---------------- Kernel 2: WMMA GEMM + grouped top-k routing ----------------
__global__ __launch_bounds__(WAVES * 32, 1)
void router_kernel(const float* __restrict__ x,
                   const unsigned short* __restrict__ wbf,
                   const float* __restrict__ bias,
                   int* __restrict__ out_idx,
                   float* __restrict__ out_w) {
    __shared__ __align__(16) unsigned char smem[SMEM_BYTES];
    unsigned short* lds_b = (unsigned short*)smem;                  // 256 x STRIDE
    unsigned short* lds_a = (unsigned short*)(smem + LDS_B_BYTES);  //  32 x STRIDE

    const int tid  = threadIdx.x;
    const int wave = tid >> 5;
    const int lane = tid & 31;
    const int lrow = lane & 15;
    const int lhi  = lane >> 4;                 // 0 or 1
    const int koff = lhi * 8;
    const int blkTok = blockIdx.x * TOK_PER_BLK;

    // wave owns M-tiles 0..1 (all 32 tokens) x N-tiles wave*4 .. wave*4+3
    v8f acc[8];                                 // acc[m*4+n]
    {
        v8f z = {0.f, 0.f, 0.f, 0.f, 0.f, 0.f, 0.f, 0.f};
        #pragma unroll
        for (int t = 0; t < 8; ++t) acc[t] = z;
    }

    for (int kc = 0; kc < KDIM; kc += KSTAGE) {
        __syncthreads();
        // Stage B: 256 x 64 bf16 (32 KB), 16 x b128 per thread
        #pragma unroll
        for (int j = 0; j < 16; ++j) {
            int u   = j * 128 + tid;            // 16B unit, 0..2047
            int row = u >> 3;                   // expert 0..255
            int q   = u & 7;
            const uint4 v = *(const uint4*)(wbf + (size_t)row * KDIM + kc + q * 8);
            *(uint4*)(&lds_b[row * STRIDE + q * 8]) = v;
        }
        // Stage A: 32 x 64 fp32 -> bf16 (4 KB), 4 x float4 per thread
        #pragma unroll
        for (int j = 0; j < 4; ++j) {
            int u   = j * 128 + tid;            // float4 unit, 0..511
            int row = u >> 4;                   // token row 0..31
            int c   = u & 15;
            float4 f = *(const float4*)(x + (size_t)(blkTok + row) * KDIM + kc + c * 4);
            v4bf pk;
            pk[0] = (__bf16)f.x; pk[1] = (__bf16)f.y;
            pk[2] = (__bf16)f.z; pk[3] = (__bf16)f.w;
            *(uint2*)(&lds_a[row * STRIDE + c * 4]) = __builtin_bit_cast(uint2, pk);
        }
        __syncthreads();
        __builtin_prefetch(x + (size_t)(blkTok + (tid & 31)) * KDIM + kc + KSTAGE, 0, 0);

        #pragma unroll
        for (int ks = 0; ks < KSTAGE; ks += 32) {
            const int off = ks + koff;
            v8u afr[2];
            #pragma unroll
            for (int m = 0; m < 2; ++m) afr[m] = loadFrag(lds_a, m, lrow, off);
            #pragma unroll
            for (int n = 0; n < 4; ++n) {
                v8u bfr = loadFrag(lds_b, wave * 4 + n, lrow, off);
                v16bf bv = __builtin_bit_cast(v16bf, bfr);
                #pragma unroll
                for (int m = 0; m < 2; ++m) {
                    acc[m * 4 + n] = __builtin_amdgcn_wmma_f32_16x16x32_bf16(
                        false, __builtin_bit_cast(v16bf, afr[m]), false, bv,
                        (short)0, acc[m * 4 + n], false, false);
                }
            }
        }
    }

    // --------- Epilogue: 32 tokens x 256 logits through reused LDS arena ---------
    float* buf = (float*)smem;                  // 32 x 257 fp32 = 32.9 KB
    __syncthreads();                            // GEMM users of arena done
    // C layout: value(lane, vgpr r) = [M = r + 8*(lane>>4)][N = lane&15]
    #pragma unroll
    for (int m = 0; m < 2; ++m) {
        #pragma unroll
        for (int n = 0; n < 4; ++n) {
            #pragma unroll
            for (int r = 0; r < 8; ++r) {
                buf[(m * 16 + r + 8 * lhi) * 257 + (wave * 64 + n * 16 + lrow)]
                    = acc[m * 4 + n][r];
            }
        }
    }
    __syncthreads();

    if (lane < 8) {
        int trow  = wave * 8 + lane;            // 0..31
        int token = blkTok + trow;
        const float* lp = &buf[trow * 257];
        // pass 1: per-group sum of top-2 (scores + bias)
        float gs[NGROUPS];
        #pragma unroll
        for (int g = 0; g < NGROUPS; ++g) {
            float m1 = -3.0e38f, m2 = -3.0e38f;
            for (int j = 0; j < GROUP_SZ; ++j) {
                int e = g * GROUP_SZ + j;
                float s  = 1.0f / (1.0f + __expf(-lp[e]));
                float sc = s + bias[e];
                if (sc > m1)      { m2 = m1; m1 = sc; }
                else if (sc > m2) { m2 = sc; }
            }
            gs[g] = m1 + m2;
        }
        // top-4 groups (ties: lowest index first, matching lax.top_k)
        unsigned gmask = 0;
        for (int r4 = 0; r4 < TOPK_GRP; ++r4) {
            float best = -3.0e38f; int bi = 0;
            #pragma unroll
            for (int g = 0; g < NGROUPS; ++g) {
                bool taken = (gmask >> g) & 1u;
                if (!taken && gs[g] > best) { best = gs[g]; bi = g; }
            }
            gmask |= (1u << bi);
        }
        // top-8 experts over tmp_scores (= sc if group selected, else 0.0)
        float tv[TOP_K]; int ti[TOP_K];
        #pragma unroll
        for (int i = 0; i < TOP_K; ++i) { tv[i] = -3.0e38f; ti[i] = 0; }
        for (int e = 0; e < N_EXP; ++e) {
            int g = e >> 5;
            float s   = 1.0f / (1.0f + __expf(-lp[e]));
            float sc  = s + bias[e];
            float val = ((gmask >> g) & 1u) ? sc : 0.0f;
            if (val > tv[TOP_K - 1]) {
                #pragma unroll
                for (int p = TOP_K - 1; p >= 0; --p) {
                    float prev  = (p > 0) ? tv[p - 1] : 3.4e38f;
                    int   prevI = (p > 0) ? ti[p - 1] : 0;
                    bool shift = val > prev;
                    bool here  = !shift && (val > tv[p]);
                    if (shift)     { tv[p] = prev; ti[p] = prevI; }
                    else if (here) { tv[p] = val;  ti[p] = e;     }
                }
            }
        }
        // weights = raw sigmoid scores at selected experts, normalized * 2.5
        float wsum = 1e-20f;
        float wv[TOP_K];
        #pragma unroll
        for (int i = 0; i < TOP_K; ++i) {
            float s = 1.0f / (1.0f + __expf(-lp[ti[i]]));
            wv[i] = s; wsum += s;
        }
        float scl = SCALING / wsum;
        #pragma unroll
        for (int i = 0; i < TOP_K; ++i) {
            out_idx[token * TOP_K + i] = ti[i];
            out_w[token * TOP_K + i]   = wv[i] * scl;
        }
    }
}

extern "C" void kernel_launch(void* const* d_in, const int* in_sizes, int n_in,
                              void* d_out, int out_size, void* d_ws, size_t ws_size,
                              hipStream_t stream) {
    const float* x    = (const float*)d_in[0];   // [8192, 7168]
    const float* gw   = (const float*)d_in[1];   // [256, 7168]
    const float* bias = (const float*)d_in[2];   // [256]

    unsigned short* wbf = (unsigned short*)d_ws; // 256*7168 bf16 = 3.67 MB
    int*   out_idx = (int*)d_out;                        // first 8192*8 ints
    float* out_w   = (float*)d_out + N_TOK * TOP_K;      // next 8192*8 floats

    // Convert gate_w to bf16: 1,835,008 elems / 4 per thread / 256 thr = 1792 blocks
    convert_w_kernel<<<(N_EXP * KDIM) / (256 * 4), 256, 0, stream>>>(gw, wbf);

    // 8192 tokens / 32 per block = 256 blocks, 128 threads (4 waves)
    router_kernel<<<N_TOK / TOK_PER_BLK, WAVES * 32, 0, stream>>>(
        x, wbf, bias, out_idx, out_w);
}
